// UnconsciousnessFlow_53455162966585
// MI455X (gfx1250) — matrix operand order repaired and checked by reading, not compiled, not run on hardware
//
#include <hip/hip_runtime.h>

// ---------------------------------------------------------------------------
// CDNA5 (gfx1250, wave32) relational-GNN step.
// v_wmma_f32_16x16x32_bf16 for all GEMMs; weight packs staged into LDS via
// global_load_async_to_lds_b128 (ASYNCcnt) and reused by persistent blocks.
//
// d_in order (jax pytree, dict keys sorted: cond_W < gate < out_W < proj):
//   0 hidden 1 edges_y 2 rel_table 3 ent_table
//   4..17  msg_params: cond_W, g0.W,g0.b, g1.W,g1.b, g2.W,g2.b, out_W,
//                      p0.W,p0.b, p1.W,p1.b, p2.W,p2.b
//   18..31 hid_params (same layout)   32 selected_edges(500000,6) int32
//
// d_ws: node_sum f32[50000*64] | node_cnt f32[50000] |
//       bf16 transposed (N-major) weight packs: msg[73728] hid[73728]
// ---------------------------------------------------------------------------

typedef __attribute__((ext_vector_type(16))) __bf16 v16bf;
typedef __attribute__((ext_vector_type(8)))  float  v8f;

#define N_NODES 50000
#define N_EDGES 500000

// unpadded global bf16 pack offsets (elements)
#define OFF_COND 0
#define OFF_P0   4096
#define OFF_G0   12288
#define OFF_P1   20480
#define OFF_G1   28672
#define OFF_P2   36864
#define OFF_G2   49152
#define OFF_OUT  61440
#define PACK_ELEMS 73728

// padded LDS pack offsets (elements); row pitch = K + 8
#define WP_COND 0       // 64 x 72
#define WP_P0   4608    // 64 x 136
#define WP_G0   13312
#define WP_P1   22016
#define WP_G1   30720
#define WP_P2   39424   // 64 x 200
#define WP_G2   52224
#define WP_OUT  65024
#define WPACK_PAD_ELEMS 77824
#define WPACK_BYTES (WPACK_PAD_ELEMS * 2)          // 155648

#define XPITCH 72     // 32x64 input block row pitch (bf16)
#define OPITCH 208    // 32x192 outs block row pitch (bf16)

// per-wave scratch: X(3x32x72 bf16)=13824B | OUT(32x208 bf16)=13312B | CB(32x64 f32)=8192B
#define WAVE_BYTES 35328
#define SMEM_BYTES (WPACK_BYTES + 4 * WAVE_BYTES)  // 296960 (< 320KB WGP limit)

__device__ __forceinline__ __bf16 f2bf(float f) {
  unsigned u = __builtin_bit_cast(unsigned, f);
  unsigned r = u + 0x7FFFu + ((u >> 16) & 1u);
  unsigned short h = (unsigned short)(r >> 16);
  return __builtin_bit_cast(__bf16, h);
}
// division-free activations: v_exp + v_rcp, no IEEE-div expansion
__device__ __forceinline__ float fast_tanh(float x) {
  x = fminf(fmaxf(x, -15.f), 15.f);
  float e = __expf(2.f * x);                      // e^(2x)
  return 1.f - 2.f * __builtin_amdgcn_rcpf(e + 1.f);
}
__device__ __forceinline__ float fast_sigm(float x) {
  return __builtin_amdgcn_rcpf(1.f + __expf(-x));
}

__device__ __forceinline__ v8f wmma_bf16(v16bf a, v16bf b, v8f c) {
  return __builtin_amdgcn_wmma_f32_16x16x32_bf16(false, a, false, b, (short)0, c,
                                                 false, false);
}

// A fragment (16x32 bf16) from LDS row-major block (base = row0 of the 16-row tile).
__device__ __forceinline__ v16bf afrag(const __bf16* blk, int pitch, int cbase, int lane) {
  int m   = lane & 15;
  int klo = cbase + ((lane >> 4) << 3);
  const __bf16* p = blk + m * pitch + klo;
  union { v16bf v; uint4 q[2]; } u;
  u.q[0] = *(const uint4*)(p);
  u.q[1] = *(const uint4*)(p + 16);
  return u.v;
}
// B fragment (32x16 bf16) from LDS N-major padded weights, row pitch KP.
__device__ __forceinline__ v16bf bfragL(const __bf16* Wt, int KP, int cbase, int nt, int lane) {
  int n  = (nt << 4) + (lane & 15);
  int kb = cbase + ((lane >> 4) << 3);
  const __bf16* p = Wt + n * KP + kb;
  union { v16bf v; uint4 q[2]; } u;
  u.q[0] = *(const uint4*)(p);
  u.q[1] = *(const uint4*)(p + 16);
  return u.v;
}

// 32-row (2 M-tiles) x 64-col GEMM chunk: A from LDS, B from LDS weight pack.
// A fragments held across the 4 N-tiles (B streams), acc[nt][mt].
template <int NKC>
__device__ __forceinline__ void gemm32(const __bf16* const* blks, const int* cbs,
                                       int pitch, const __bf16* W, int KP, int lane,
                                       v8f (*acc)[2]) {
  const v8f z = {0.f, 0.f, 0.f, 0.f, 0.f, 0.f, 0.f, 0.f};
#pragma unroll
  for (int nt = 0; nt < 4; ++nt) { acc[nt][0] = z; acc[nt][1] = z; }
#pragma unroll
  for (int kc = 0; kc < NKC; ++kc) {
    v16bf a0 = afrag(blks[kc], pitch, cbs[kc], lane);
    v16bf a1 = afrag(blks[kc] + 16 * pitch, pitch, cbs[kc], lane);
#pragma unroll
    for (int nt = 0; nt < 4; ++nt) {
      v16bf b = bfragL(W, KP, kc * 32, nt, lane);
      acc[nt][0] = wmma_bf16(a0, b, acc[nt][0]);
      acc[nt][1] = wmma_bf16(a1, b, acc[nt][1]);
    }
  }
}

// stage bf16 weight pack (global, unpadded) -> LDS (padded rows) via async copy
__device__ __forceinline__ void stage_weights(const __bf16* __restrict__ wsrc,
                                              char* smem, int tid) {
  struct M { int srcOff; int K; int dstOff; };
  const M ms[8] = {{OFF_COND, 64, WP_COND}, {OFF_P0, 128, WP_P0}, {OFF_G0, 128, WP_G0},
                   {OFF_P1, 128, WP_P1},    {OFF_G1, 128, WP_G1}, {OFF_P2, 192, WP_P2},
                   {OFF_G2, 192, WP_G2},    {OFF_OUT, 192, WP_OUT}};
#pragma unroll
  for (int m = 0; m < 8; ++m) {
    int K = ms[m].K, KP = K + 8;
    int cpr = K >> 3;              // 16B chunks per row
    int total = 64 * cpr;          // multiple of 128 -> EXEC stays full
    for (int c = tid; c < total; c += 128) {
      int n = c / cpr, off = (c - n * cpr) << 3;
      const __bf16* g = wsrc + ms[m].srcOff + n * K + off;
      unsigned ldsaddr =
          (unsigned)(unsigned long long)(smem + (ms[m].dstOff + n * KP + off) * 2);
      asm volatile("global_load_async_to_lds_b128 %0, %1, off"
                   :: "v"(ldsaddr), "v"(g) : "memory");
    }
  }
  asm volatile("s_wait_asynccnt 0x0" ::: "memory");
  __syncthreads();
}

// apply_F on 32 rows. X = 3 blocks of 32x64 bf16 (cond = block 0).
// Final tanh results land in CB (32x64 f32, reused from the cond buffer).
__device__ __forceinline__ void apply_f_32(
    __bf16* X, __bf16* OUTb, float* CB, const __bf16* W,
    const float* pb0, const float* pb1, const float* pb2,
    const float* gb0, const float* gb1, const float* gb2,
    int p00, int p01, int p10, int p11, int square, int lane) {
  const __bf16* bl[3] = {X, X + 32 * XPITCH, X + 64 * XPITCH};
  const float* pB[3] = {pb0, pb1, pb2};
  const float* gB[3] = {gb0, gb1, gb2};
  int nn = lane & 15, mb = (lane >> 4) << 3;
  v8f acc[4][2];

  { // cond = tanh(X0 @ cond_W) (^2 for hidden phase)
    const __bf16* blks[2] = {bl[0], bl[0]};
    int cbs[2] = {0, 32};
    gemm32<2>(blks, cbs, XPITCH, W + WP_COND, 72, lane, acc);
    for (int nt = 0; nt < 4; ++nt)
      for (int mt = 0; mt < 2; ++mt)
        for (int v = 0; v < 8; ++v) {
          float t = fast_tanh(acc[nt][mt][v]);
          CB[(mt * 16 + mb + v) * 64 + nt * 16 + nn] = square ? t * t : t;
        }
  }

  const int WPp[3] = {WP_P0, WP_P1, WP_P2}, WPg[3] = {WP_G0, WP_G1, WP_G2};
  int ids0[2] = {p00, p01}, ids1[2] = {p10, p11}, ids2[3] = {0, 1, 2};

  for (int i = 0; i < 3; ++i) {
    int L = (i == 2) ? 3 : 2;
    const int* ids = (i == 0) ? ids0 : (i == 1) ? ids1 : ids2;
    const __bf16* blks[6]; int cbs[6];
    for (int kc = 0; kc < 2 * L; ++kc) { blks[kc] = bl[ids[kc >> 1]]; cbs[kc] = (kc & 1) * 32; }
    int KP = 64 * L + 8;

    // proj pass -> OUT = bf16(tanh(xWp + bp))
    if (L == 2) gemm32<4>(blks, cbs, XPITCH, W + WPp[i], KP, lane, acc);
    else        gemm32<6>(blks, cbs, XPITCH, W + WPp[i], KP, lane, acc);
    for (int nt = 0; nt < 4; ++nt) {
      int n = nt * 16 + nn; float bp = pB[i][n];
      for (int mt = 0; mt < 2; ++mt)
        for (int v = 0; v < 8; ++v)
          OUTb[(mt * 16 + mb + v) * OPITCH + i * 64 + n] =
              f2bf(fast_tanh(acc[nt][mt][v] + bp));
    }
    // gate pass -> OUT *= sigmoid(xWg + bg) * c
    if (L == 2) gemm32<4>(blks, cbs, XPITCH, W + WPg[i], KP, lane, acc);
    else        gemm32<6>(blks, cbs, XPITCH, W + WPg[i], KP, lane, acc);
    for (int nt = 0; nt < 4; ++nt) {
      int n = nt * 16 + nn; float bg = gB[i][n];
      for (int mt = 0; mt < 2; ++mt)
        for (int v = 0; v < 8; ++v) {
          int row = mt * 16 + mb + v;
          float t = (float)OUTb[row * OPITCH + i * 64 + n];
          float o = t * fast_sigm(acc[nt][mt][v] + bg) * CB[row * 64 + n];
          OUTb[row * OPITCH + i * 64 + n] = f2bf(o);
        }
    }
  }

  { // final: tanh(outs(32x192) @ out_W) -> CB
    const __bf16* blks[6]; int cbs[6];
    for (int kc = 0; kc < 6; ++kc) { blks[kc] = OUTb; cbs[kc] = kc * 32; }
    gemm32<6>(blks, cbs, OPITCH, W + WP_OUT, 200, lane, acc);
    for (int nt = 0; nt < 4; ++nt)
      for (int mt = 0; mt < 2; ++mt)
        for (int v = 0; v < 8; ++v)
          CB[(mt * 16 + mb + v) * 64 + nt * 16 + nn] = fast_tanh(acc[nt][mt][v]);
  }
}

// ---- prep: f32 (Kdim x 64) -> bf16 transposed (64 x Kdim) ----
__global__ void prep_kernel(const float* __restrict__ src, __bf16* __restrict__ dst, int Kdim) {
  int i = blockIdx.x * blockDim.x + threadIdx.x;
  if (i < 64 * Kdim) {
    int n = i / Kdim, k = i - n * Kdim;
    dst[i] = f2bf(src[k * 64 + n]);
  }
}
__global__ void zero_kernel(float* __restrict__ p, int n) {
  int i = blockIdx.x * blockDim.x + threadIdx.x;
  if (i < n) p[i] = 0.f;
}
__global__ void count_kernel(const int* __restrict__ edges, float* __restrict__ cnt, int nE) {
  int e = blockIdx.x * blockDim.x + threadIdx.x;
  if (e < nE) atomicAdd(cnt + edges[6 * (long)e + 2], 1.0f);
}

// ---- phase 1: per-edge messages + atomic segment-sum over vj ----
__global__ void __launch_bounds__(128) msg_kernel(
    const float* __restrict__ hidden, const float* __restrict__ edges_y,
    const float* __restrict__ rel_table, const int* __restrict__ edges,
    const __bf16* __restrict__ wpack,
    const float* pb0, const float* pb1, const float* pb2,
    const float* gb0, const float* gb1, const float* gb2,
    float* __restrict__ node_sum, int nwt) {
  extern __shared__ char smem[];
  int tid = threadIdx.x, wid = tid >> 5, lane = tid & 31;
  stage_weights(wpack, smem, tid);
  char* wv = smem + WPACK_BYTES + wid * WAVE_BYTES;
  __bf16* X    = (__bf16*)wv;
  __bf16* OUTb = (__bf16*)(wv + 13824);
  float*  CB   = (float*)(wv + 27136);
  const __bf16* W = (const __bf16*)smem;

  for (int wt = blockIdx.x * 4 + wid; wt < nwt; wt += gridDim.x * 4) {
    long ebase = (long)wt * 32;
#pragma unroll 4
    for (int r = 0; r < 32; ++r) {
      long e = ebase + r;
      int vi = edges[6 * e + 1], vj = edges[6 * e + 2], rl = edges[6 * e + 3];
      float2 a = *(const float2*)(hidden    + (long)vi * 64 + lane * 2);
      float2 b = *(const float2*)(rel_table + (long)rl * 64 + lane * 2);
      float2 c = *(const float2*)(hidden    + (long)vj * 64 + lane * 2);
      X[r * XPITCH + 2 * lane] = f2bf(a.x);          X[r * XPITCH + 2 * lane + 1] = f2bf(a.y);
      X[(32 + r) * XPITCH + 2 * lane] = f2bf(b.x);   X[(32 + r) * XPITCH + 2 * lane + 1] = f2bf(b.y);
      X[(64 + r) * XPITCH + 2 * lane] = f2bf(c.x);   X[(64 + r) * XPITCH + 2 * lane + 1] = f2bf(c.y);
    }
    apply_f_32(X, OUTb, CB, W, pb0, pb1, pb2, gb0, gb1, gb2,
               /*pairs*/ 0, 1, 1, 2, /*square*/ 0, lane);
#pragma unroll 4
    for (int r = 0; r < 32; ++r) {
      long e = ebase + r;
      int vj = edges[6 * e + 2];
      float ey = edges_y[e];
      float2 m = *(const float2*)(CB + r * 64 + 2 * lane);
      atomicAdd(node_sum + (long)vj * 64 + 2 * lane,     m.x * ey);
      atomicAdd(node_sum + (long)vj * 64 + 2 * lane + 1, m.y * ey);
    }
  }
}

// ---- phase 2: hidden update ----
__global__ void __launch_bounds__(128) hid_kernel(
    const float* __restrict__ hidden, const float* __restrict__ ent_table,
    const float* __restrict__ node_sum, const float* __restrict__ node_cnt,
    const __bf16* __restrict__ wpack,
    const float* pb0, const float* pb1, const float* pb2,
    const float* gb0, const float* gb1, const float* gb2,
    float* __restrict__ out, int nwt) {
  extern __shared__ char smem[];
  int tid = threadIdx.x, wid = tid >> 5, lane = tid & 31;
  stage_weights(wpack, smem, tid);
  char* wv = smem + WPACK_BYTES + wid * WAVE_BYTES;
  __bf16* X    = (__bf16*)wv;
  __bf16* OUTb = (__bf16*)(wv + 13824);
  float*  CB   = (float*)(wv + 27136);
  const __bf16* W = (const __bf16*)smem;

  for (int wt = blockIdx.x * 4 + wid; wt < nwt; wt += gridDim.x * 4) {
    long nbase = (long)wt * 32;
#pragma unroll 4
    for (int r = 0; r < 32; ++r) {
      long nd = nbase + r;
      if (nd > N_NODES - 1) nd = N_NODES - 1;   // clamp tail (rows unused)
      float inv = __builtin_amdgcn_rcpf(fmaxf(node_cnt[nd], 1.0f));
      float2 s  = *(const float2*)(node_sum  + nd * 64 + lane * 2);
      float2 h  = *(const float2*)(hidden    + nd * 64 + lane * 2);
      float2 en = *(const float2*)(ent_table + nd * 64 + lane * 2);
      X[r * XPITCH + 2 * lane] = f2bf(s.x * inv);      X[r * XPITCH + 2 * lane + 1] = f2bf(s.y * inv);
      X[(32 + r) * XPITCH + 2 * lane] = f2bf(h.x);     X[(32 + r) * XPITCH + 2 * lane + 1] = f2bf(h.y);
      X[(64 + r) * XPITCH + 2 * lane] = f2bf(en.x);    X[(64 + r) * XPITCH + 2 * lane + 1] = f2bf(en.y);
    }
    apply_f_32(X, OUTb, CB, W, pb0, pb1, pb2, gb0, gb1, gb2,
               /*pairs*/ 0, 1, 0, 2, /*square*/ 1, lane);
#pragma unroll 4
    for (int r = 0; r < 32; ++r) {
      long nd = nbase + r;
      if (nd >= N_NODES) break;
      float2 h = *(const float2*)(hidden + nd * 64 + lane * 2);
      float2 m = *(const float2*)(CB + r * 64 + 2 * lane);
      float2 o; o.x = h.x + m.x; o.y = h.y + m.y;
      *(float2*)(out + nd * 64 + lane * 2) = o;
    }
  }
}

extern "C" void kernel_launch(void* const* d_in, const int* in_sizes, int n_in,
                              void* d_out, int out_size, void* d_ws, size_t ws_size,
                              hipStream_t stream) {
  const float* hidden    = (const float*)d_in[0];
  const float* edges_y   = (const float*)d_in[1];
  const float* rel_table = (const float*)d_in[2];
  const float* ent_table = (const float*)d_in[3];
  const int*   edges     = (const int*)d_in[32];

  float*  node_sum = (float*)d_ws;
  float*  node_cnt = node_sum + (size_t)N_NODES * 64;
  __bf16* wbase    = (__bf16*)(node_cnt + N_NODES);
  __bf16* wmsg     = wbase;
  __bf16* whid     = wbase + PACK_ELEMS;

  struct PD { int src; int K; int off; };
  const PD pd[16] = {
      {4, 64, OFF_COND}, {12, 128, OFF_P0}, {5, 128, OFF_G0},
      {14, 128, OFF_P1}, {7, 128, OFF_G1},  {16, 192, OFF_P2},
      {9, 192, OFF_G2},  {11, 192, OFF_OUT},
      {18, 64, PACK_ELEMS + OFF_COND}, {26, 128, PACK_ELEMS + OFF_P0},
      {19, 128, PACK_ELEMS + OFF_G0},  {28, 128, PACK_ELEMS + OFF_P1},
      {21, 128, PACK_ELEMS + OFF_G1},  {30, 192, PACK_ELEMS + OFF_P2},
      {23, 192, PACK_ELEMS + OFF_G2},  {25, 192, PACK_ELEMS + OFF_OUT}};
  for (int i = 0; i < 16; ++i) {
    int tot = 64 * pd[i].K;
    prep_kernel<<<(tot + 255) / 256, 256, 0, stream>>>(
        (const float*)d_in[pd[i].src], wbase + pd[i].off, pd[i].K);
  }

  int zn = N_NODES * 64 + N_NODES;
  zero_kernel<<<(zn + 255) / 256, 256, 0, stream>>>(node_sum, zn);
  count_kernel<<<(N_EDGES + 255) / 256, 256, 0, stream>>>(edges, node_cnt, N_EDGES);

  int mwt = N_EDGES / 32;                  // 15625 wave-tiles of 32 edges
  int mblk = 512;                          // persistent blocks, grid-stride
  msg_kernel<<<mblk, 128, SMEM_BYTES, stream>>>(
      hidden, edges_y, rel_table, edges, wmsg,
      (const float*)d_in[13], (const float*)d_in[15], (const float*)d_in[17],
      (const float*)d_in[6], (const float*)d_in[8], (const float*)d_in[10],
      node_sum, mwt);

  int hwt = (N_NODES + 31) / 32;           // 1563 (tail clamped)
  int hblk = (hwt + 3) / 4;
  hid_kernel<<<hblk, 128, SMEM_BYTES, stream>>>(
      hidden, ent_table, node_sum, node_cnt, whid,
      (const float*)d_in[27], (const float*)d_in[29], (const float*)d_in[31],
      (const float*)d_in[20], (const float*)d_in[22], (const float*)d_in[24],
      (float*)d_out, hwt);
}
